// RoINet_12214886989943
// MI455X (gfx1250) — compile-verified, matrix operand order checked by reading, not run on hardware
//
#include <hip/hip_runtime.h>
#include <hip/hip_bf16.h>

// ---------------- problem constants (match reference) ----------------
#define B_    8
#define N_    4000
#define C_    80
#define NCLS  81          // C+1
#define KPRE  2048
#define DET_  100
#define CAP   4096        // candidate capacity before final top-K sort
#define CLIPV 4.135166556742356f   // log(1000/16)
#define CLS_OFF 4096.0f

typedef unsigned int       u32;
typedef unsigned long long u64;

typedef __attribute__((ext_vector_type(16))) _Float16 v16h;
typedef __attribute__((ext_vector_type(8)))  float    v8f;
typedef int v4i_g __attribute__((vector_size(16)));   // matches builtin param type

#if defined(__HIP_DEVICE_COMPILE__) && defined(__gfx1250__) && \
    __has_builtin(__builtin_amdgcn_global_load_async_to_lds_b128)
#define ASYNC_OK 1
#define AS1 __attribute__((address_space(1)))
#define AS3 __attribute__((address_space(3)))
#endif

// ---------------- shared device helpers ----------------
__device__ __forceinline__ float4 decode_box(float4 d, float4 p) {
    float w  = p.z - p.x;
    float h  = p.w - p.y;
    float cx = p.x + 0.5f * w;
    float cy = p.y + 0.5f * h;
    float dx = d.x * 0.1f;                       // /10
    float dy = d.y * 0.1f;                       // /10
    float dw = fminf(d.z * 0.2f, CLIPV);         // /5, clipped
    float dh = fminf(d.w * 0.2f, CLIPV);
    float pcx = dx * w + cx;
    float pcy = dy * h + cy;
    float pw  = __expf(dw) * w;
    float ph  = __expf(dh) * h;
    return make_float4(pcx - 0.5f * pw, pcy - 0.5f * ph,
                       pcx + 0.5f * pw, pcy + 0.5f * ph);
}

// In-LDS bitonic sort, descending by u64 key. M = power-of-two size, T = blockDim.
template <int M, int T>
__device__ __forceinline__ void bitonic_desc(u64* keys, int tid) {
    for (int size = 2; size <= M; size <<= 1) {
        for (int stride = size >> 1; stride > 0; stride >>= 1) {
            __syncthreads();
            for (int i = tid; i < M; i += T) {
                int j = i ^ stride;
                if (j > i) {
                    u64 a = keys[i], b = keys[j];
                    bool up = ((i & size) == 0);
                    if (up ? (a < b) : (a > b)) { keys[i] = b; keys[j] = a; }
                }
            }
        }
    }
    __syncthreads();
}

// ---------------- Kernel 1: fused softmax (WMMA row-sums) + decode + mask ----------------
// One wave handles 16 proposals. A-fragment layout per CDNA5 ISA 7.12.2:
//   lanes 0-15 : M=lane,    VGPR0-3 -> K=0..7,  VGPR4-7 -> K=16..23
//   lanes16-31 : M=lane-16, VGPR0-3 -> K=8..15, VGPR4-7 -> K=24..31
// B = ones so every column of D equals the row sum of exp().
__global__ __launch_bounds__(256) void k_softmax_decode(
    const float* __restrict__ logits, const float* __restrict__ deltas,
    const float* __restrict__ props, float* __restrict__ scores_all)
{
    const int wave  = blockIdx.x * 8 + (threadIdx.x >> 5);
    const int lane  = threadIdx.x & 31;
    const int wslot = threadIdx.x >> 5;
    __shared__ float s_sum[8 * 16];

    const int b    = wave / (N_ / 16);
    const int grp  = wave % (N_ / 16);
    const int n0   = grp * 16;
    const int half = lane >> 4;
    const int m    = lane & 15;
    const int row  = n0 + m;

    const float* lg = logits + ((size_t)b * N_ + row) * NCLS;

    float rowmax = -1e30f;
    for (int c = 0; c < NCLS; ++c) rowmax = fmaxf(rowmax, lg[c]);

    v16h bones;
    for (int i = 0; i < 16; ++i) bones[i] = (_Float16)1.0f;

    v8f acc = {};
    for (int ck = 0; ck < NCLS; ck += 32) {
        v16h a;
        for (int i = 0; i < 8; ++i) {
            int c1 = ck + half * 8 + i;
            a[i]     = (c1 < NCLS) ? (_Float16)__expf(lg[c1] - rowmax) : (_Float16)0.0f;
            int c2 = ck + 16 + half * 8 + i;
            a[8 + i] = (c2 < NCLS) ? (_Float16)__expf(lg[c2] - rowmax) : (_Float16)0.0f;
        }
        acc = __builtin_amdgcn_wmma_f32_16x16x32_f16(
            false, a, false, bones, (short)0, acc, false, false);
    }
    // acc[j] on lanes 0-15 holds rowsum(j); on lanes 16-31 holds rowsum(j+8)
    if (m == 0) {
        for (int j = 0; j < 8; ++j) s_sum[wslot * 16 + half * 8 + j] = acc[j];
    }
    __syncthreads();

    // Pass 2: 16 rows x 80 foreground classes -> masked scores
    for (int t = lane; t < 16 * C_; t += 32) {
        int r = t / C_;
        int c = t - r * C_;          // 0..79, label = c+1
        int n = n0 + r;
        size_t rowoff = (size_t)b * N_ + n;
        float lv = logits[rowoff * NCLS + (c + 1)];
        float rm = __shfl(rowmax, r, 32);          // lane r computed row r's max
        float s  = __expf(lv - rm) / s_sum[wslot * 16 + r];
        float4 pr = ((const float4*)props)[rowoff];
        float4 dl = ((const float4*)deltas)[rowoff * NCLS + (c + 1)];
        float4 bx = decode_box(dl, pr);
        float area = (bx.w - bx.y) * (bx.z - bx.x);
        bool valid = (s > 0.01f) && (area > 0.1f);
        scores_all[(size_t)b * (N_ * C_) + (size_t)n * C_ + c] = valid ? s : -1.0f;
    }
}

// ---------------- Kernels 2-6: exact top-2048 via 2-level radix threshold ----------------
__global__ __launch_bounds__(256) void k_hist1(const float* __restrict__ sa,
                                               u32* __restrict__ hist1) {
    size_t i = (size_t)blockIdx.x * 256 + threadIdx.x;
    if (i >= (size_t)B_ * N_ * C_) return;
    float s = sa[i];
    if (s > 0.0f) {
        int b = (int)(i / (N_ * C_));
        atomicAdd(&hist1[b * 1024 + (__float_as_uint(s) >> 20)], 1u);
    }
}

__global__ void k_thresh1(const u32* __restrict__ hist1, u32* __restrict__ thr) {
    if (threadIdx.x) return;
    int b = blockIdx.x;
    const u32* h = hist1 + b * 1024;
    u32 run = 0, high = 0; int t1 = 0; bool found = false;
    for (int t = 1023; t >= 1; --t) {
        u32 nr = run + h[t];
        if (!found && nr >= KPRE) { t1 = t; high = run; found = true; }
        run = nr;
    }
    if (!found) { t1 = 0; high = run; }      // fewer than KPRE positives
    thr[b * 4 + 0] = (u32)t1;
    thr[b * 4 + 2] = high;                   // count strictly above bin t1
}

__global__ __launch_bounds__(256) void k_hist2(const float* __restrict__ sa,
                                               const u32* __restrict__ thr,
                                               u32* __restrict__ hist2) {
    size_t i = (size_t)blockIdx.x * 256 + threadIdx.x;
    if (i >= (size_t)B_ * N_ * C_) return;
    float s = sa[i];
    if (s <= 0.0f) return;
    int b = (int)(i / (N_ * C_));
    u32 bits = __float_as_uint(s);
    if ((bits >> 20) == thr[b * 4]) atomicAdd(&hist2[b * 256 + ((bits >> 12) & 0xFF)], 1u);
}

__global__ void k_thresh2(const u32* __restrict__ hist2, u32* __restrict__ thr) {
    if (threadIdx.x) return;
    int b = blockIdx.x;
    u32 t1 = thr[b * 4], high = thr[b * 4 + 2];
    u32 rem = KPRE - high;
    const u32* h = hist2 + b * 256;
    u32 run = 0; int t2 = 0; bool found = false;
    for (int s = 255; s >= 1; --s) {
        run += h[s];
        if (!found && run >= rem) { t2 = s; found = true; }
    }
    thr[b * 4 + 1] = (t1 << 8) | (u32)t2;    // 20-bit threshold on (bits>>12)
}

__global__ __launch_bounds__(256) void k_compact(const float* __restrict__ sa,
                                                 const u32* __restrict__ thr,
                                                 u32* __restrict__ cnt,
                                                 u64* __restrict__ cands) {
    size_t i = (size_t)blockIdx.x * 256 + threadIdx.x;
    if (i >= (size_t)B_ * N_ * C_) return;
    float s = sa[i];
    if (s <= 0.0f) return;
    int b = (int)(i / (N_ * C_));
    u32 bits = __float_as_uint(s);
    if ((bits >> 12) < thr[b * 4 + 1]) return;
    u32 pos = atomicAdd(&cnt[b], 1u);
    if (pos < CAP) {
        u32 idx = (u32)(i - (size_t)b * (N_ * C_));
        // key: score bits high (positive floats sort like uints), ~idx low
        // so equal scores break ties toward the LOWEST index (matches top_k).
        cands[(size_t)b * CAP + pos] = ((u64)bits << 32) | (u32)(~idx);
    }
}

__global__ __launch_bounds__(512) void k_sort_cand(const u64* __restrict__ cands,
                                                   float* __restrict__ tsc,
                                                   u32* __restrict__ tidx) {
    const int b = blockIdx.x, tid = threadIdx.x;
    __shared__ u64 keys[CAP];                 // 32 KB LDS
    for (int k = tid; k < CAP; k += 512) keys[k] = cands[(size_t)b * CAP + k];
    bitonic_desc<CAP, 512>(keys, tid);
    for (int k = tid; k < KPRE; k += 512) {
        u64 key = keys[k];
        float s; u32 idx;
        if (key == 0ull) { s = -1.0f; idx = 0u; }       // padding slot
        else { s = __uint_as_float((u32)(key >> 32)); idx = ~(u32)(key & 0xFFFFFFFFull); }
        tsc[b * KPRE + k]  = s;
        tidx[b * KPRE + k] = idx;
    }
}

// ---------------- Kernel 8: re-decode selected boxes (+class offset for NMS) ----------------
__global__ __launch_bounds__(256) void k_gather(const u32* __restrict__ tidx,
                                                const float* __restrict__ deltas,
                                                const float* __restrict__ props,
                                                float* __restrict__ boxes,
                                                float* __restrict__ nbox) {
    int gid = blockIdx.x * 256 + threadIdx.x;
    if (gid >= B_ * KPRE) return;
    int b = gid / KPRE;
    u32 idx = tidx[gid];
    int n = (int)(idx / C_), c = (int)(idx % C_);
    size_t rowoff = (size_t)b * N_ + n;
    float4 pr = ((const float4*)props)[rowoff];
    float4 dl = ((const float4*)deltas)[rowoff * NCLS + (c + 1)];
    float4 bx = decode_box(dl, pr);
    float off = (float)(c + 1) * CLS_OFF;
    ((float4*)boxes)[gid] = bx;
    ((float4*)nbox)[gid]  = make_float4(bx.x + off, bx.y + off, bx.z + off, bx.w + off);
}

// ---------------- Kernel 9: suppression bitmask (async LDS staging of box tile) ----------------
__global__ __launch_bounds__(256) void k_supmat(const float* __restrict__ nbox,
                                                u32* __restrict__ sup) {
    const int b = blockIdx.y;
    const int tid = threadIdx.x;
    __shared__ float4 sb[KPRE];               // 32 KB box tile in LDS
    const float4* g = (const float4*)nbox + (size_t)b * KPRE;
#if ASYNC_OK
    for (int k = tid; k < KPRE; k += 256)
        __builtin_amdgcn_global_load_async_to_lds_b128(
            (AS1 v4i_g*)(g + k), (AS3 v4i_g*)(&sb[k]), 0, 0);
    asm volatile("s_wait_asynccnt 0" ::: "memory");
#else
    for (int k = tid; k < KPRE; k += 256) sb[k] = g[k];
#endif
    __syncthreads();

    const int r = blockIdx.x * 256 + tid;
    float4 rb = sb[r];
    float ra = fmaxf(rb.z - rb.x, 0.f) * fmaxf(rb.w - rb.y, 0.f);
    u32* outrow = sup + ((size_t)b * KPRE + r) * 64;
    for (int w = 0; w < 64; ++w) {
        u32 word = 0u;
        if (w * 32 + 31 > r) {
            for (int bi = 0; bi < 32; ++bi) {
                int j = w * 32 + bi;
                if (j > r) {
                    float4 ob = sb[j];
                    float oa = fmaxf(ob.z - ob.x, 0.f) * fmaxf(ob.w - ob.y, 0.f);
                    float lx = fmaxf(rb.x, ob.x), ly = fmaxf(rb.y, ob.y);
                    float rx = fminf(rb.z, ob.z), ry = fminf(rb.w, ob.w);
                    float inter = fmaxf(rx - lx, 0.f) * fmaxf(ry - ly, 0.f);
                    float iou = inter / fmaxf(ra + oa - inter, 1e-9f);
                    if (iou > 0.5f) word |= (1u << bi);
                }
            }
        }
        outrow[w] = word;
    }
}

// ---------------- Kernel 10: serial greedy scan, one wave32 per image ----------------
// lane l owns removed-mask words l (cols 0..1023) and 32+l (cols 1024..2047).
// Next row is prefetched unconditionally to pipeline the dependent L2 loads.
__global__ __launch_bounds__(32) void k_nms_scan(const u32* __restrict__ sup,
                                                 const float* __restrict__ tsc,
                                                 float* __restrict__ fsc) {
    const int b = blockIdx.x;
    const int l = threadIdx.x;
    const u32* S = sup + (size_t)b * KPRE * 64;
    u32 remv0 = 0, remv1 = 0;
    u32 w0 = S[l], w1 = S[32 + l];
    for (int i = 0; i < KPRE; ++i) {
        u32 nw0 = 0, nw1 = 0;
        if (i + 1 < KPRE) {
            __builtin_prefetch(&S[(size_t)(i + 2) * 64], 0, 0);
            nw0 = S[(size_t)(i + 1) * 64 + l];
            nw1 = S[(size_t)(i + 1) * 64 + 32 + l];
        }
        int wsel = i >> 5;                              // word index 0..63 (uniform)
        u32 cur  = (wsel < 32) ? remv0 : remv1;
        u32 word = (u32)__shfl((int)cur, wsel & 31, 32);
        float sc = tsc[b * KPRE + i];
        bool alive = (((word >> (i & 31)) & 1u) == 0u) && (sc > 0.0f);
        if (alive) { remv0 |= w0; remv1 |= w1; }
        if (l == 0) fsc[b * KPRE + i] = alive ? sc : -1.0f;
        w0 = nw0; w1 = nw1;
    }
}

// ---------------- Kernel 11: final top-100 sort + output ----------------
__global__ __launch_bounds__(256) void k_final(const float* __restrict__ fsc,
                                               const float* __restrict__ boxes,
                                               const u32* __restrict__ tidx,
                                               float* __restrict__ out) {
    const int b = blockIdx.x, tid = threadIdx.x;
    __shared__ u64 keys[KPRE];                // 16 KB
    for (int k = tid; k < KPRE; k += 256) {
        float s = fsc[b * KPRE + k];
        u32 u = __float_as_uint(s);
        u ^= (u >> 31) ? 0xFFFFFFFFu : 0x80000000u;   // total order incl. negatives
        keys[k] = ((u64)u << 32) | (u32)(~k);
    }
    bitonic_desc<KPRE, 256>(keys, tid);
    for (int k = tid; k < DET_; k += 256) {
        u32 idx = ~(u32)(keys[k] & 0xFFFFFFFFull);    // 0..2047
        float  s  = fsc[b * KPRE + idx];
        float4 bx = ((const float4*)boxes)[b * KPRE + idx];
        float* ob = out + ((size_t)b * DET_ + k) * 4;
        ob[0] = bx.x; ob[1] = bx.y; ob[2] = bx.z; ob[3] = bx.w;
        out[(size_t)B_ * DET_ * 4 + b * DET_ + k] = s;
        u32 oi = tidx[b * KPRE + idx];
        out[(size_t)B_ * DET_ * 4 + (size_t)B_ * DET_ + b * DET_ + k] =
            (float)(oi % C_ + 1);
    }
}

// ---------------- host side ----------------
extern "C" void kernel_launch(void* const* d_in, const int* in_sizes, int n_in,
                              void* d_out, int out_size, void* d_ws, size_t ws_size,
                              hipStream_t stream) {
    const float* logits = (const float*)d_in[0];   // [B,N,81]
    const float* deltas = (const float*)d_in[1];   // [B,N,324]
    const float* props  = (const float*)d_in[2];   // [B,N,4]
    float* out = (float*)d_out;

    char* w = (char*)d_ws;
    size_t off = 0;
    auto A = [&](size_t bytes) -> char* {
        char* p = w + off;
        off = (off + bytes + 255) & ~(size_t)255;
        return p;
    };
    float* scores_all = (float*)A((size_t)B_ * N_ * C_ * 4);   // 10.24 MB
    u32*   hist1      = (u32*)  A((size_t)B_ * 1024 * 4);
    u32*   hist2      = (u32*)  A((size_t)B_ * 256 * 4);
    u32*   thr        = (u32*)  A((size_t)B_ * 4 * 4);
    u32*   cnt        = (u32*)  A((size_t)B_ * 4);
    u64*   cands      = (u64*)  A((size_t)B_ * CAP * 8);
    float* tsc        = (float*)A((size_t)B_ * KPRE * 4);
    u32*   tidx       = (u32*)  A((size_t)B_ * KPRE * 4);
    float* boxes      = (float*)A((size_t)B_ * KPRE * 16);
    float* nbox       = (float*)A((size_t)B_ * KPRE * 16);
    u32*   sup        = (u32*)  A((size_t)B_ * KPRE * 64 * 4); // 4 MB
    float* fsc        = (float*)A((size_t)B_ * KPRE * 4);

    // zero hist1..cands (contiguous region) — counters, histograms, sort padding
    size_t zbytes = (size_t)((char*)(cands + (size_t)B_ * CAP) - (char*)hist1);
    (void)hipMemsetAsync(hist1, 0, zbytes, stream);

    const int totBlocks = (B_ * N_ * C_ + 255) / 256;          // 10000
    k_softmax_decode<<<(B_ * (N_ / 16)) / 8, 256, 0, stream>>>(logits, deltas, props, scores_all);
    k_hist1  <<<totBlocks, 256, 0, stream>>>(scores_all, hist1);
    k_thresh1<<<B_, 32, 0, stream>>>(hist1, thr);
    k_hist2  <<<totBlocks, 256, 0, stream>>>(scores_all, thr, hist2);
    k_thresh2<<<B_, 32, 0, stream>>>(hist2, thr);
    k_compact<<<totBlocks, 256, 0, stream>>>(scores_all, thr, cnt, cands);
    k_sort_cand<<<B_, 512, 0, stream>>>(cands, tsc, tidx);
    k_gather <<<(B_ * KPRE + 255) / 256, 256, 0, stream>>>(tidx, deltas, props, boxes, nbox);
    k_supmat <<<dim3(KPRE / 256, B_), 256, 0, stream>>>(nbox, sup);
    k_nms_scan<<<B_, 32, 0, stream>>>(sup, tsc, fsc);
    k_final  <<<B_, 256, 0, stream>>>(fsc, boxes, tidx, out);

    (void)in_sizes; (void)n_in; (void)out_size; (void)ws_size;
}
